// BicameralAttention_18769007084193
// MI455X (gfx1250) — matrix operand order repaired and checked by reading, not compiled, not run on hardware
//
#include <hip/hip_runtime.h>

// Problem constants
#define BB 4
#define LL 2048
#define DD 1024
#define HH 16
#define HDD 64
#define NROWS_SIDE 65536  // B*8*L rows per attention side

typedef __attribute__((ext_vector_type(16))) __bf16 v16bf;
typedef __attribute__((ext_vector_type(8)))  float  v8f;

union BFrag {
  v16bf v;
  unsigned int w[8];
};

__device__ __forceinline__ unsigned short f2bf(float f) {
  unsigned int u = __float_as_uint(f);
  u += 0x7FFFu + ((u >> 16) & 1u);   // round-to-nearest-even
  return (unsigned short)(u >> 16);
}
__device__ __forceinline__ unsigned int pack2(float lo, float hi) {
  return (unsigned int)f2bf(lo) | ((unsigned int)f2bf(hi) << 16);
}
__device__ __forceinline__ v8f wmma_bf16(BFrag a, BFrag b, v8f c) {
  return __builtin_amdgcn_wmma_f32_16x16x32_bf16(false, a.v, false, b.v,
                                                 (short)0, c, false, false);
}
__device__ __forceinline__ v8f vzero8() {
  v8f z = {0.f, 0.f, 0.f, 0.f, 0.f, 0.f, 0.f, 0.f};
  return z;
}

// ---------------------------------------------------------------------------
// Kernel 1: QKV projection.  x[8192x1024] @ qkv_w[1024x3072] + qkv_b.
// Block tile 128x128, BK=32, 8 waves (4x2), wave tile 32x64 (2x4 wmma tiles).
// Epilogue: tt (Q/K/V select), h, b are wave-uniform -> one scalar branch.
// ---------------------------------------------------------------------------
__global__ __launch_bounds__(256) void qkv_gemm_kernel(
    const float* __restrict__ x, const float* __restrict__ w,
    const float* __restrict__ bias,
    unsigned short* __restrict__ Qd, unsigned short* __restrict__ Kd,
    unsigned short* __restrict__ Vt) {
  __shared__ unsigned short As[128][40];  // [m][k] bf16, padded
  __shared__ unsigned short Bs[128][40];  // [n][k] bf16 (transposed), padded

  const int t    = threadIdx.x;
  const int lane = t & 31;
  const int wv   = t >> 5;
  const int wm   = wv >> 1;  // 0..3  (M direction)
  const int wn   = wv & 1;   // 0..1  (N direction)
  const int l16  = lane & 15;
  const int hs   = lane >> 4;
  const int mBase = blockIdx.y * 128;
  const int nBase = blockIdx.x * 128;

  v8f acc[2][4];
#pragma unroll
  for (int i = 0; i < 2; ++i)
#pragma unroll
    for (int j = 0; j < 4; ++j) acc[i][j] = vzero8();

  const int arow = t >> 1, acg = (t & 1) * 16;
  const int bk = t >> 3, bng = (t & 7) * 16;

  for (int k0 = 0; k0 < DD; k0 += 32) {
    {  // stage A: 128x32 fp32 -> bf16 LDS
      const float* src = x + (size_t)(mBase + arow) * DD + k0 + acg;
      unsigned int* dst = (unsigned int*)&As[arow][acg];
#pragma unroll
      for (int i = 0; i < 4; ++i) {
        float4 f = *(const float4*)(src + 4 * i);
        dst[2 * i + 0] = pack2(f.x, f.y);
        dst[2 * i + 1] = pack2(f.z, f.w);
      }
    }
    {  // stage B: 32x128 fp32 -> bf16 LDS (transposed store)
      const float* src = w + (size_t)(k0 + bk) * 3072 + nBase + bng;
#pragma unroll
      for (int i = 0; i < 4; ++i) {
        float4 f = *(const float4*)(src + 4 * i);
        Bs[bng + 4 * i + 0][bk] = f2bf(f.x);
        Bs[bng + 4 * i + 1][bk] = f2bf(f.y);
        Bs[bng + 4 * i + 2][bk] = f2bf(f.z);
        Bs[bng + 4 * i + 3][bk] = f2bf(f.w);
      }
    }
    __syncthreads();

    BFrag aF[2], bF[4];
    const int kb = hs * 8;
#pragma unroll
    for (int mt = 0; mt < 2; ++mt) {
      const int r = wm * 32 + mt * 16 + l16;
#pragma unroll
      for (int v = 0; v < 8; ++v) {
        const int kc = (v < 4 ? 2 * v : 16 + 2 * (v - 4)) + kb;
        aF[mt].w[v] = *(const unsigned int*)&As[r][kc];
      }
    }
    const int kb2 = hs * 16;
#pragma unroll
    for (int nt = 0; nt < 4; ++nt) {
      const int c = wn * 64 + nt * 16 + l16;
#pragma unroll
      for (int v = 0; v < 8; ++v)
        bF[nt].w[v] = *(const unsigned int*)&Bs[c][(v < 4 ? 2 * v : 16 + 2 * (v - 4)) + kb];
    }
#pragma unroll
    for (int mt = 0; mt < 2; ++mt)
#pragma unroll
      for (int nt = 0; nt < 4; ++nt)
        acc[mt][nt] = wmma_bf16(aF[mt], bF[nt], acc[mt][nt]);
    __syncthreads();
  }

  // Epilogue: all of b, h, tt are wave-uniform.
  const int n0 = nBase + wn * 64;                         // uniform in wave
  const int tt = __builtin_amdgcn_readfirstlane(n0 >> 10);      // 0/1/2
  const int h  = __builtin_amdgcn_readfirstlane((n0 >> 6) & 15);
  const int b  = __builtin_amdgcn_readfirstlane(mBase >> 11);
  const int bh = b * HH + h;
  const int lRow0 = (mBase & 2047) + wm * 32 + hs * 8;

  float bv[4];
#pragma unroll
  for (int nt = 0; nt < 4; ++nt) bv[nt] = bias[n0 + nt * 16 + l16];

  if (tt < 2) {  // Q or K: (B,H,L,64), contiguous hd per lane
    unsigned short* dst = (tt == 0 ? Qd : Kd) + (size_t)bh * LL * HDD;
#pragma unroll
    for (int mt = 0; mt < 2; ++mt)
#pragma unroll
      for (int v = 0; v < 8; ++v) {
        const int l = lRow0 + mt * 16 + v;
        unsigned short* row = dst + (size_t)l * HDD;
#pragma unroll
        for (int nt = 0; nt < 4; ++nt)
          row[nt * 16 + l16] = f2bf(acc[mt][nt][v] + bv[nt]);
      }
  } else {  // V transposed: (B,H,64,L)
    unsigned short* dst = Vt + (size_t)bh * HDD * LL;
#pragma unroll
    for (int mt = 0; mt < 2; ++mt)
#pragma unroll
      for (int v = 0; v < 8; ++v) {
        const int l = lRow0 + mt * 16 + v;
#pragma unroll
        for (int nt = 0; nt < 4; ++nt)
          dst[(size_t)(nt * 16 + l16) * LL + l] = f2bf(acc[mt][nt][v] + bv[nt]);
      }
  }
}

// ---------------------------------------------------------------------------
// Kernel 2: scores S = (Q K^T)*scale/8, causal mask on local heads, plus
// online softmax stats (running max + rescaled sum) per row -> stats buffers.
// Skips fully-masked tiles entirely (except one alignment tile of -inf so the
// AV kernel's 32-wide steps always read initialized data).
// ---------------------------------------------------------------------------
__global__ __launch_bounds__(256) void scores_kernel(
    const unsigned short* __restrict__ Qd, const unsigned short* __restrict__ Kd,
    float* __restrict__ attnL, float* __restrict__ attnG,
    const float* __restrict__ lscale, const float* __restrict__ gscale,
    float* __restrict__ statsM, float* __restrict__ statsS) {
  const int lane = threadIdx.x & 31;
  const int wv = threadIdx.x >> 5;
  const int qt = blockIdx.x * 8 + wv;   // 0..127
  const int bh = blockIdx.y;            // b*16+h
  const int h = bh & 15, b = bh >> 4;
  const bool isLocal = (h < 8);
  const int hh = isLocal ? h : h - 8;
  float* S = isLocal ? attnL + (size_t)(b * 8 + hh) * LL * LL
                     : attnG + (size_t)(b * 8 + hh) * LL * LL;
  const size_t statBase =
      (isLocal ? 0 : (size_t)NROWS_SIDE) + (size_t)(b * 8 + hh) * LL;
  const float scale = (isLocal ? lscale[0] : gscale[0]) * 0.125f;
  const int l16 = lane & 15, hs = lane >> 4;
  const int qBase = qt * 16;
  const int kb = hs * 8, kb2 = hs * 16;
  const float NEG_INF = -__builtin_inff();

  // Q fragments (hd 0..31 and 32..63), reused for all key tiles
  BFrag qF[2];
  const unsigned short* Qrow = Qd + ((size_t)bh * LL + qBase + l16) * HDD;
#pragma unroll
  for (int v = 0; v < 8; ++v) {
    const int hd = (v < 4 ? 2 * v : 16 + 2 * (v - 4)) + kb;
    qF[0].w[v] = *(const unsigned int*)(Qrow + hd);
    qF[1].w[v] = *(const unsigned int*)(Qrow + hd + 32);
  }

  float rm[8], rs[8];
#pragma unroll
  for (int v = 0; v < 8; ++v) {
    rm[v] = NEG_INF;
    rs[v] = 0.f;
  }

  const int ktEnd = isLocal ? (qt + 1) : 128;
  for (int kt = 0; kt < ktEnd; ++kt) {
    const int kBase = kt * 16;
    BFrag kF[2];
    const unsigned short* Krow = Kd + ((size_t)bh * LL + kBase + l16) * HDD;
#pragma unroll
    for (int v = 0; v < 8; ++v) {
      const int hd = (v < 4 ? 2 * v : 16 + 2 * (v - 4)) + kb;
      kF[0].w[v] = *(const unsigned int*)(Krow + hd);
      kF[1].w[v] = *(const unsigned int*)(Krow + hd + 32);
    }
    v8f c = vzero8();
    c = wmma_bf16(qF[0], kF[0], c);
    c = wmma_bf16(qF[1], kF[1], c);
#pragma unroll
    for (int v = 0; v < 8; ++v) {
      const int q = qBase + v + hs * 8;
      const int k = kBase + l16;
      float s = c[v] * scale;
      if (isLocal && k > q) s = NEG_INF;
      S[(size_t)q * LL + k] = s;
      // online softmax update for row q (16 lanes of this half-wave)
      float tmax = s;
#pragma unroll
      for (int off = 8; off >= 1; off >>= 1)
        tmax = fmaxf(tmax, __shfl_xor(tmax, off, 16));
      const float mNew = fmaxf(rm[v], tmax);
      const float corr = __expf(rm[v] - mNew);  // 0 on first tile
      float p = __expf(s - mNew);
#pragma unroll
      for (int off = 8; off >= 1; off >>= 1) p += __shfl_xor(p, off, 16);
      rs[v] = rs[v] * corr + p;
      rm[v] = mNew;
    }
  }

  // Alignment tile of -inf so AV's 32-wide step has initialized data.
  if (isLocal && (qt & 1) == 0) {
    const int kBase = (qt + 1) * 16;
#pragma unroll
    for (int v = 0; v < 8; ++v) {
      const int q = qBase + v + hs * 8;
      S[(size_t)q * LL + kBase + l16] = NEG_INF;
    }
  }

  // lane 0 of each half-wave writes stats for its 8 rows
  if (l16 == 0) {
#pragma unroll
    for (int v = 0; v < 8; ++v) {
      const size_t r = statBase + qBase + v + hs * 8;
      statsM[r] = rm[v];
      statsS[r] = rs[v];
    }
  }
}

// ---------------------------------------------------------------------------
// Kernel 3: fused normalize + O = attn @ V.  Reads raw S, computes
// p = exp(s-m)/sum in registers, writes p (the required attn output) back,
// and feeds p (bf16) to WMMA against Vt.  Zero-fills the causal upper
// triangle instead of computing it.
// ---------------------------------------------------------------------------
__global__ __launch_bounds__(256) void av_kernel(
    float* attnL, float* attnG,
    const float* __restrict__ statsM, const float* __restrict__ statsS,
    const unsigned short* __restrict__ Vt, unsigned short* __restrict__ hout) {
  const int lane = threadIdx.x & 31;
  const int wv = threadIdx.x >> 5;
  const int qt = blockIdx.x * 8 + wv;
  const int bh = blockIdx.y;
  const int h = bh & 15, b = bh >> 4;
  const bool isLocal = (h < 8);
  const int hh = isLocal ? h : h - 8;
  float* P = isLocal ? attnL + (size_t)(b * 8 + hh) * LL * LL
                     : attnG + (size_t)(b * 8 + hh) * LL * LL;
  const size_t statBase =
      (isLocal ? 0 : (size_t)NROWS_SIDE) + (size_t)(b * 8 + hh) * LL;
  const int l16 = lane & 15, hs = lane >> 4;
  const int qBase = qt * 16;
  const int kb = hs * 8, kb2 = hs * 16;
  float* Arow = P + (size_t)(qBase + l16) * LL;

  // per-lane row normalization constants (A-frag row = qBase + l16)
  const float mRow = statsM[statBase + qBase + l16];
  const float invS = 1.0f / statsS[statBase + qBase + l16];

  v8f acc[4];
#pragma unroll
  for (int nt = 0; nt < 4; ++nt) acc[nt] = vzero8();

  const int nsteps = isLocal ? ((qBase + 47) >> 5) : 64;  // causal early exit
  for (int s = 0; s < nsteps; ++s) {
    const int ks = s * 32;
    BFrag aF;
#pragma unroll
    for (int v = 0; v < 8; ++v) {
      const int kc = ks + kb2 + 2 * v;
      float2 f = *(const float2*)(Arow + kc);
      const float p0 = __expf(f.x - mRow) * invS;
      const float p1 = __expf(f.y - mRow) * invS;
      float2 pw;
      pw.x = p0;
      pw.y = p1;
      *(float2*)(Arow + kc) = pw;  // final attention probabilities
      aF.w[v] = pack2(p0, p1);
    }
#pragma unroll
    for (int nt = 0; nt < 4; ++nt) {
      BFrag bF;
      const unsigned short* Vrow =
          Vt + ((size_t)bh * HDD + nt * 16 + l16) * LL + ks + kb2;
#pragma unroll
      for (int v = 0; v < 8; ++v)
        bF.w[v] = *(const unsigned int*)(Vrow + 2 * v);
      acc[nt] = wmma_bf16(aF, bF, acc[nt]);
    }
  }

  // zero-fill masked (never-computed) columns of the prob output
  if (isLocal) {
    for (int c0 = 32 * nsteps; c0 < LL; c0 += 32) {
#pragma unroll
      for (int r = 0; r < 16; ++r)
        P[(size_t)(qBase + r) * LL + c0 + lane] = 0.f;
    }
  }

  // store bf16 head-concatenated intermediate (B, L, H*HD)
#pragma unroll
  for (int nt = 0; nt < 4; ++nt) {
#pragma unroll
    for (int v = 0; v < 8; ++v) {
      const int q = qBase + v + hs * 8;
      const int hd = nt * 16 + l16;
      hout[((size_t)(b * LL + q)) * DD + h * HDD + hd] = f2bf(acc[nt][v]);
    }
  }
}

// ---------------------------------------------------------------------------
// Kernel 4: out = hout(bf16) @ out_w + out_b  -> fp32 d_out.
// ---------------------------------------------------------------------------
__global__ __launch_bounds__(256) void outproj_kernel(
    const unsigned short* __restrict__ hout, const float* __restrict__ w,
    const float* __restrict__ bias, float* __restrict__ out) {
  __shared__ unsigned short As[128][40];
  __shared__ unsigned short Bs[128][40];

  const int t = threadIdx.x;
  const int lane = t & 31;
  const int wv = t >> 5;
  const int wm = wv >> 1;
  const int wn = wv & 1;
  const int l16 = lane & 15;
  const int hs = lane >> 4;
  const int mBase = blockIdx.y * 128;
  const int nBase = blockIdx.x * 128;

  v8f acc[2][4];
#pragma unroll
  for (int i = 0; i < 2; ++i)
#pragma unroll
    for (int j = 0; j < 4; ++j) acc[i][j] = vzero8();

  const int arow = t >> 1, acg = (t & 1) * 16;
  const int bk = t >> 3, bng = (t & 7) * 16;

  for (int k0 = 0; k0 < DD; k0 += 32) {
    {  // stage A: bf16 straight copy
      const unsigned short* src = hout + (size_t)(mBase + arow) * DD + k0 + acg;
      *(uint4*)&As[arow][acg] = ((const uint4*)src)[0];
      *(uint4*)&As[arow][acg + 8] = ((const uint4*)src)[1];
    }
    {  // stage B: fp32 -> bf16 transposed
      const float* src = w + (size_t)(k0 + bk) * DD + nBase + bng;
#pragma unroll
      for (int i = 0; i < 4; ++i) {
        float4 f = *(const float4*)(src + 4 * i);
        Bs[bng + 4 * i + 0][bk] = f2bf(f.x);
        Bs[bng + 4 * i + 1][bk] = f2bf(f.y);
        Bs[bng + 4 * i + 2][bk] = f2bf(f.z);
        Bs[bng + 4 * i + 3][bk] = f2bf(f.w);
      }
    }
    __syncthreads();

    BFrag aF[2], bF[4];
    const int kb = hs * 8;
#pragma unroll
    for (int mt = 0; mt < 2; ++mt) {
      const int r = wm * 32 + mt * 16 + l16;
#pragma unroll
      for (int v = 0; v < 8; ++v) {
        const int kc = (v < 4 ? 2 * v : 16 + 2 * (v - 4)) + kb;
        aF[mt].w[v] = *(const unsigned int*)&As[r][kc];
      }
    }
    const int kb2 = hs * 16;
#pragma unroll
    for (int nt = 0; nt < 4; ++nt) {
      const int c = wn * 64 + nt * 16 + l16;
#pragma unroll
      for (int v = 0; v < 8; ++v)
        bF[nt].w[v] = *(const unsigned int*)&Bs[c][(v < 4 ? 2 * v : 16 + 2 * (v - 4)) + kb];
    }
#pragma unroll
    for (int mt = 0; mt < 2; ++mt)
#pragma unroll
      for (int nt = 0; nt < 4; ++nt)
        acc[mt][nt] = wmma_bf16(aF[mt], bF[nt], acc[mt][nt]);
    __syncthreads();
  }

#pragma unroll
  for (int mt = 0; mt < 2; ++mt) {
#pragma unroll
    for (int nt = 0; nt < 4; ++nt) {
      const int n = nBase + wn * 64 + nt * 16 + l16;
      const float bv = bias[n];
#pragma unroll
      for (int v = 0; v < 8; ++v) {
        const int m = mBase + wm * 32 + mt * 16 + v + hs * 8;
        out[(size_t)m * DD + n] = acc[mt][nt][v] + bv;
      }
    }
  }
}

// ---------------------------------------------------------------------------
extern "C" void kernel_launch(void* const* d_in, const int* in_sizes, int n_in,
                              void* d_out, int out_size, void* d_ws,
                              size_t ws_size, hipStream_t stream) {
  const float* x      = (const float*)d_in[0];
  const float* qkv_w  = (const float*)d_in[1];
  const float* qkv_b  = (const float*)d_in[2];
  const float* out_w  = (const float*)d_in[3];
  const float* out_b  = (const float*)d_in[4];
  const float* lscale = (const float*)d_in[5];
  const float* gscale = (const float*)d_in[6];

  float* out = (float*)d_out;
  float* attnL = out + (size_t)BB * LL * DD;
  float* attnG = attnL + (size_t)BB * 8 * LL * LL;

  const size_t qkv_elems = (size_t)BB * HH * LL * HDD;  // 8388608
  unsigned short* Qd = (unsigned short*)d_ws;
  unsigned short* Kd = Qd + qkv_elems;
  unsigned short* Vt = Kd + qkv_elems;
  unsigned short* hout = Vt + qkv_elems;
  float* statsM = (float*)(hout + qkv_elems);
  float* statsS = statsM + (size_t)2 * NROWS_SIDE;

  qkv_gemm_kernel<<<dim3(3072 / 128, (BB * LL) / 128), 256, 0, stream>>>(
      x, qkv_w, qkv_b, Qd, Kd, Vt);
  scores_kernel<<<dim3(LL / 16 / 8, BB * HH), 256, 0, stream>>>(
      Qd, Kd, attnL, attnG, lscale, gscale, statsM, statsS);
  av_kernel<<<dim3(LL / 16 / 8, BB * HH), 256, 0, stream>>>(
      attnL, attnG, statsM, statsS, Vt, hout);
  outproj_kernel<<<dim3(DD / 128, (BB * LL) / 128), 256, 0, stream>>>(
      hout, out_w, out_b, out);
}
